// Planner_32461362823245
// MI455X (gfx1250) — matrix operand order repaired and checked by reading, not compiled, not run on hardware
//
#include <hip/hip_runtime.h>
#include <hip/hip_bf16.h>
#include <math.h>

// ---------------------------------------------------------------------------
// VIN-style planner on MI455X (gfx1250, wave32, WMMA).
// Heavy contractions (conv 128->128, LSTM h@W_hh^T) run on
// v_wmma_f32_16x16x32_f16 (f16 operands, f32 accumulate).
// Fragment loads are 2x b128 per operand half (contiguous per ISA layout),
// padding handled with branchless value selects (no exec-mask predication).
// ---------------------------------------------------------------------------

typedef __attribute__((ext_vector_type(16))) _Float16 v16h;
typedef __attribute__((ext_vector_type(8)))  float    v8f;
typedef __attribute__((ext_vector_type(4)))  unsigned v4u;

#define BB    16
#define MM    32
#define LH    128
#define NPIX  (BB * MM * MM)      // 16384
#define NGATE (4 * LH)            // 512

union Frag { v16h v; v4u q[2]; };

// A fragment (16x32 f16): lane row = lane&15; halves [kbase+8*hi .. +7] and
// [kbase+16+8*hi .. +7] -> two contiguous 16B chunks.
__device__ __forceinline__ v16h load_frag_a(const _Float16* __restrict__ row,
                                            int kbase, int hi) {
    Frag f;
    f.q[0] = *(const v4u*)(row + kbase + 8 * hi);
    f.q[1] = *(const v4u*)(row + kbase + 16 + 8 * hi);
    return f.v;
}

__device__ __forceinline__ v16h load_frag_a_masked(const _Float16* __restrict__ row,
                                                   int kbase, int hi, bool valid) {
    v4u q0 = *(const v4u*)(row + kbase + 8 * hi);          // always-safe (clamped)
    v4u q1 = *(const v4u*)(row + kbase + 16 + 8 * hi);
    v4u z  = (v4u)0u;
    Frag f;
    f.q[0] = valid ? q0 : z;                               // v_cndmask, no branch
    f.q[1] = valid ? q1 : z;
    return f.v;
}

// B fragment (32x16 f16): lane col = lane&15; halves [kbase+8*hi .. +7] and
// [kbase+16+8*hi .. +7] -> two contiguous 16B chunks.
__device__ __forceinline__ v16h load_frag_b(const _Float16* __restrict__ row,
                                            int kbase, int hi) {
    Frag f;
    f.q[0] = *(const v4u*)(row + kbase + 8 * hi);
    f.q[1] = *(const v4u*)(row + kbase + 16 + 8 * hi);
    return f.v;
}

// --------------------------- prep kernels ----------------------------------

__global__ __launch_bounds__(256) void f32_to_f16_kernel(
    const float* __restrict__ src, _Float16* __restrict__ dst, int n) {
    int i = blockIdx.x * 256 + threadIdx.x;
    if (i < n) dst[i] = (_Float16)src[i];
}

// W (3,3,128,128) HWIO  ->  Wt[tap][co][ci] f16 (B-fragment friendly)
__global__ __launch_bounds__(256) void conv_w_transpose_kernel(
    const float* __restrict__ W, _Float16* __restrict__ Wt) {
    int t = blockIdx.x * 256 + threadIdx.x;          // 9*128*128 = 147456
    int co  = t & 127;
    int ci  = (t >> 7) & 127;
    int tap = t >> 14;
    Wt[tap * (LH * LH) + co * LH + ci] =
        (_Float16)W[(tap * LH + ci) * LH + co];
}

// ----------------------- input conv (Cin=2 -> 128) -------------------------

__global__ __launch_bounds__(256) void conv_hid_kernel(
    const float* __restrict__ md, const float* __restrict__ gl,
    const float* __restrict__ W, const float* __restrict__ bias,
    _Float16* __restrict__ hid16) {
    int t  = blockIdx.x * 256 + threadIdx.x;         // NPIX*128 threads
    int co = t & 127;
    int p  = t >> 7;
    int x = p & 31, y = (p >> 5) & 31, b = p >> 10;
    float acc = bias[co];
#pragma unroll
    for (int tap = 0; tap < 9; ++tap) {
        int dy = tap / 3 - 1, dx = tap % 3 - 1;
        int y2 = y + dy, x2 = x + dx;
        bool valid = (y2 >= 0) & (y2 < MM) & (x2 >= 0) & (x2 < MM);
        int pp = valid ? ((b * MM + y2) * MM + x2) : 0;
        float mv = md[pp], gv = gl[pp];
        mv = valid ? mv : 0.f;
        gv = valid ? gv : 0.f;
        acc += mv * W[(tap * 2 + 0) * LH + co];
        acc += gv * W[(tap * 2 + 1) * LH + co];
    }
    hid16[t] = (_Float16)acc;
}

// ------------------- 3x3 conv 128->128 via WMMA GEMM -----------------------
// grid: 1024 pixel blocks (b,y,xblk), block: 256 threads = 8 waves,
// each wave computes one 16-wide C_out tile, 9 taps x 4 k-chunks WMMAs.

__global__ __launch_bounds__(256) void conv3x3_128_wmma_kernel(
    const _Float16* __restrict__ in,    // (NPIX, 128) f16
    const _Float16* __restrict__ Wt,    // (9, 128co, 128ci) f16
    const float* __restrict__ bias,     // (128)
    float* __restrict__ of32,           // optional (NPIX,128)
    _Float16* __restrict__ of16) {      // optional (NPIX,128)
    int pb   = blockIdx.x;
    int wave = threadIdx.x >> 5;
    int lane = threadIdx.x & 31;
    int r  = lane & 15;
    int hi = lane >> 4;
    int xblk = pb & 1, y = (pb >> 1) & 31, b = pb >> 6;
    int x = xblk * 16 + r;               // A-row pixel x for this lane
    int cobase = wave * 16;

    v8f acc = {};
#pragma unroll
    for (int tap = 0; tap < 9; ++tap) {
        int dy = tap / 3 - 1, dx = tap % 3 - 1;
        int y2 = y + dy, x2 = x + dx;
        bool valid = (y2 >= 0) & (y2 < MM) & (x2 >= 0) & (x2 < MM);
        int pp = valid ? ((b * MM + y2) * MM + x2) : 0;   // clamped: safe addr
        const _Float16* arow = in + (size_t)pp * LH;
        const _Float16* brow = Wt + (size_t)tap * (LH * LH)
                                  + (size_t)(cobase + r) * LH;
#pragma unroll
        for (int kc = 0; kc < 4; ++kc) {
            int kbase = kc * 32;
            v16h A  = load_frag_a_masked(arow, kbase, hi, valid);
            v16h Bf = load_frag_b(brow, kbase, hi);
            acc = __builtin_amdgcn_wmma_f32_16x16x32_f16(
                false, A, false, Bf, (short)0, acc, false, false);
        }
    }
    int col = cobase + r;
    float bv = bias[col];
#pragma unroll
    for (int v = 0; v < 8; ++v) {
        int p = pb * 16 + v + 8 * hi;    // C/D layout: row = v + 8*hi
        float val = acc[v] + bv;
        if (of32) of32[(size_t)p * LH + col] = val;
        if (of16) of16[(size_t)p * LH + col] = (_Float16)val;
    }
}

// ----------------------- per-step 128 -> 1 conv ----------------------------

__global__ __launch_bounds__(256) void conv_f_kernel(
    const _Float16* __restrict__ h16, const float* __restrict__ Wf,
    const float* __restrict__ bf, float* __restrict__ inp) {
    int p = blockIdx.x * 256 + threadIdx.x;          // NPIX threads
    int x = p & 31, y = (p >> 5) & 31, b = p >> 10;
    float acc = bf[0];
#pragma unroll
    for (int tap = 0; tap < 9; ++tap) {
        int dy = tap / 3 - 1, dx = tap % 3 - 1;
        int y2 = y + dy, x2 = x + dx;
        bool valid = (y2 >= 0) & (y2 < MM) & (x2 >= 0) & (x2 < MM);
        int pp = valid ? ((b * MM + y2) * MM + x2) : 0;
        const _Float16* hp = h16 + (size_t)pp * LH;
        const float* wp = Wf + tap * LH;
        float s = 0.f;
        for (int ci = 0; ci < LH; ++ci) s += (float)hp[ci] * wp[ci];
        acc += valid ? s : 0.f;
    }
    inp[p] = acc;
}

// ------------- fused LSTM step: WMMA gate GEMM + cell update ---------------
// grid: 1024 pixel blocks, 256 threads (8 waves). Each wave: 4 col-tiles of
// the 16x512 gate tile, K = 128 (4 WMMAs each). Gates staged in LDS.

__global__ __launch_bounds__(256) void lstm_step_wmma_kernel(
    const _Float16* __restrict__ Whh16,  // (512, 128) f16
    const float* __restrict__ inp,       // (NPIX)
    const float* __restrict__ W_ih,      // (512)
    const float* __restrict__ b_ih,      // (512)
    const float* __restrict__ b_hh,      // (512)
    float* __restrict__ c,               // (NPIX, 128) f32, in/out
    _Float16* __restrict__ h16) {        // (NPIX, 128) f16, in/out
    __shared__ float gb[16 * NGATE];     // 32 KB

    int pb   = blockIdx.x;
    int wave = threadIdx.x >> 5;
    int lane = threadIdx.x & 31;
    int r  = lane & 15;
    int hi = lane >> 4;

    // keep the shared W_hh panel warm (global_prefetch_b8)
    __builtin_prefetch(Whh16 + (size_t)(wave * 64 + lane) * LH, 0, 3);

    // A fragments: h rows pb*16 .. pb*16+15, all 4 K-chunks
    const _Float16* arow = h16 + (size_t)(pb * 16 + r) * LH;
    v16h A[4];
#pragma unroll
    for (int kc = 0; kc < 4; ++kc) A[kc] = load_frag_a(arow, kc * 32, hi);

#pragma unroll
    for (int j = 0; j < 4; ++j) {
        int colbase = (wave * 4 + j) * 16;
        const _Float16* brow = Whh16 + (size_t)(colbase + r) * LH;
        v8f acc = {};
#pragma unroll
        for (int kc = 0; kc < 4; ++kc) {
            v16h Bf = load_frag_b(brow, kc * 32, hi);
            acc = __builtin_amdgcn_wmma_f32_16x16x32_f16(
                false, A[kc], false, Bf, (short)0, acc, false, false);
        }
        int col = colbase + r;
#pragma unroll
        for (int v = 0; v < 8; ++v)
            gb[(v + 8 * hi) * NGATE + col] = acc[v];
    }
    __syncthreads();

    // elementwise LSTM update: 16 pixels x 128 channels = 2048 elems
#pragma unroll
    for (int e = 0; e < 8; ++e) {
        int lin = threadIdx.x + e * 256;
        int pix = lin >> 7;
        int ch  = lin & 127;
        int p   = pb * 16 + pix;
        float xin = inp[p];
        float gi = gb[pix * NGATE + ch]           + xin * W_ih[ch]           + b_ih[ch]           + b_hh[ch];
        float gf = gb[pix * NGATE + LH + ch]      + xin * W_ih[LH + ch]      + b_ih[LH + ch]      + b_hh[LH + ch];
        float gg = gb[pix * NGATE + 2 * LH + ch]  + xin * W_ih[2 * LH + ch]  + b_ih[2 * LH + ch]  + b_hh[2 * LH + ch];
        float go = gb[pix * NGATE + 3 * LH + ch]  + xin * W_ih[3 * LH + ch]  + b_ih[3 * LH + ch]  + b_hh[3 * LH + ch];
        float si = 1.f / (1.f + __expf(-gi));
        float sf = 1.f / (1.f + __expf(-gf));
        float so = 1.f / (1.f + __expf(-go));
        float tg = tanhf(gg);
        size_t idx = (size_t)p * LH + ch;
        float c2 = sf * c[idx] + si * tg;
        float h2 = so * tanhf(c2);
        c[idx]   = c2;
        h16[idx] = (_Float16)h2;
    }
}

// ------------------ policy (1x1 conv 128->4) + softmax ---------------------

__global__ __launch_bounds__(256) void policy_kernel(
    const _Float16* __restrict__ h16, const float* __restrict__ Wpol,
    float* __restrict__ out) {
    int p = blockIdx.x * 256 + threadIdx.x;          // NPIX threads
    const _Float16* hp = h16 + (size_t)p * LH;
    float lg[4] = {0.f, 0.f, 0.f, 0.f};
    for (int ci = 0; ci < LH; ++ci) {
        float hv = (float)hp[ci];
#pragma unroll
        for (int a = 0; a < 4; ++a) lg[a] += hv * Wpol[ci * 4 + a];
    }
    float mx = lg[0];
#pragma unroll
    for (int a = 1; a < 4; ++a) mx = fmaxf(mx, lg[a]);
    float ex[4], s = 0.f;
#pragma unroll
    for (int a = 0; a < 4; ++a) { ex[a] = __expf(lg[a] - mx); s += ex[a]; }
    float inv = 1.f / s;
    int b = p >> 10, yx = p & 1023;
#pragma unroll
    for (int a = 0; a < 4; ++a) {
        size_t o = (size_t)(b * 4 + a) * 1024 + yx;  // (B, NA, NO=1, M, M)
        out[o]                 = lg[a];              // logits
        out[(size_t)NPIX * 4 + o] = ex[a] * inv;     // probs
    }
}

// ------------------------------- launcher ----------------------------------

extern "C" void kernel_launch(void* const* d_in, const int* in_sizes, int n_in,
                              void* d_out, int out_size, void* d_ws, size_t ws_size,
                              hipStream_t stream) {
    const float* md    = (const float*)d_in[0];
    const float* gl    = (const float*)d_in[1];
    const float* W_hid = (const float*)d_in[2];
    const float* b_hid = (const float*)d_in[3];
    const float* W_h0  = (const float*)d_in[4];
    const float* b_h0  = (const float*)d_in[5];
    const float* W_c0  = (const float*)d_in[6];
    const float* b_c0  = (const float*)d_in[7];
    const float* W_f   = (const float*)d_in[8];
    const float* b_f   = (const float*)d_in[9];
    const float* W_ih  = (const float*)d_in[10];
    const float* b_ih  = (const float*)d_in[11];
    const float* W_hh  = (const float*)d_in[12];
    const float* b_hh  = (const float*)d_in[13];
    const float* W_pol = (const float*)d_in[14];

    char* ws = (char*)d_ws;
    _Float16* whh16 = (_Float16*)(ws);                    // 512*128*2   = 128 KB
    _Float16* wt_h0 = (_Float16*)(ws + 131072);           // 9*128*128*2 = 288 KB
    _Float16* wt_c0 = (_Float16*)(ws + 425984);           // 288 KB
    _Float16* hid16 = (_Float16*)(ws + 720896);           // 4 MB
    _Float16* h16   = (_Float16*)(ws + 4915200);          // 4 MB
    float*    c32   = (float*)   (ws + 9109504);          // 8 MB
    float*    inp   = (float*)   (ws + 17498112);         // 64 KB

    // weight prep
    f32_to_f16_kernel<<<256, 256, 0, stream>>>(W_hh, whh16, NGATE * LH);
    conv_w_transpose_kernel<<<576, 256, 0, stream>>>(W_h0, wt_h0);
    conv_w_transpose_kernel<<<576, 256, 0, stream>>>(W_c0, wt_c0);

    // hid = conv(X, W_hid)
    conv_hid_kernel<<<(NPIX * LH) / 256, 256, 0, stream>>>(md, gl, W_hid, b_hid, hid16);

    // h0 / c0 = conv(hid)
    conv3x3_128_wmma_kernel<<<NPIX / 16, 256, 0, stream>>>(
        hid16, wt_h0, b_h0, (float*)nullptr, h16);
    conv3x3_128_wmma_kernel<<<NPIX / 16, 256, 0, stream>>>(
        hid16, wt_c0, b_c0, c32, (_Float16*)nullptr);

    // 15 LSTM steps (K-1)
    for (int step = 0; step < 15; ++step) {
        conv_f_kernel<<<NPIX / 256, 256, 0, stream>>>(h16, W_f, b_f, inp);
        lstm_step_wmma_kernel<<<NPIX / 16, 256, 0, stream>>>(
            whh16, inp, W_ih, b_ih, b_hh, c32, h16);
    }

    // policy head + softmax
    policy_kernel<<<NPIX / 256, 256, 0, stream>>>(h16, W_pol, (float*)d_out);
}
